// MultiLevelSpikeDrivenSelfAttention_56985626083431
// MI455X (gfx1250) — matrix operand order repaired and checked by reading, not compiled, not run on hardware
//
#include <hip/hip_runtime.h>
#include <hip/hip_bf16.h>

// ---------------------------------------------------------------------------
// MultiLevelSpikeDrivenSelfAttention for MI455X (gfx1250, wave32, WMMA)
//   0) one-time f32 -> f16 conversion of x and all weights
//   1) QKV projection GEMMs (async-to-LDS double-buffered, fused neuron+thresh)
//   2) temporal avg-pool (s=3,5)
//   3) sigmoid attention per scale (k/v tiles staged to LDS once per block)
//   4) fused upsample+concat+Wo GEMM+neuron
// All GEMMs: v_wmma_f32_16x16x32_f16 (f16 in, f32 accumulate).
// Data movement: global_load_async_to_lds_b128 (ASYNCcnt) + s_wait_asynccnt.
// Steady-state loops are peeled so the hot path is branch-free:
//   stage(next) -> s_wait_asynccnt N -> barrier -> ds_loads -> wmma x4 -> barrier
// ---------------------------------------------------------------------------

typedef __attribute__((ext_vector_type(16))) _Float16 v16h;
typedef __attribute__((ext_vector_type(8)))  _Float16 v8h;
typedef __attribute__((ext_vector_type(4)))  float    v4f;
typedef __attribute__((ext_vector_type(8)))  float    v8f;

#define WMMA_F16(a, b, c) \
  __builtin_amdgcn_wmma_f32_16x16x32_f16(false, (a), false, (b), (short)0, (c), false, false)

#define B_   2
#define T_   1920
#define D_   1024
#define H_   16
#define DH_  64
#define M_   (B_ * T_)      // 3840 total rows
#define TS3  640
#define TS5  384

__device__ __forceinline__ float sigf(float x) { return 1.0f / (1.0f + __expf(-x)); }

__device__ __forceinline__ v16h cat8h(v8h lo, v8h hi) {
  return __builtin_shufflevector(lo, hi, 0, 1, 2, 3, 4, 5, 6, 7, 8, 9, 10, 11, 12, 13, 14, 15);
}

// Async copy of 16 bytes global -> LDS, tracked by ASYNCcnt.
__device__ __forceinline__ void async_b128(const _Float16* g, _Float16* l) {
  unsigned int       loff = (unsigned int)(uintptr_t)l;       // low 32b = LDS addr
  unsigned long long ga   = (unsigned long long)(uintptr_t)g;
  asm volatile("global_load_async_to_lds_b128 %0, %1, off" :: "v"(loff), "v"(ga) : "memory");
}
#define WAIT_ASYNC(n) asm volatile("s_wait_asynccnt " #n ::: "memory")

// ---------------------------------------------------------------------------
// Kernel 0: elementwise f32 -> f16 (8 elements / thread)
// ---------------------------------------------------------------------------
__global__ __launch_bounds__(256) void cvt_kernel(
    const float* __restrict__ in, _Float16* __restrict__ out, int n8)
{
  const int i = blockIdx.x * 256 + threadIdx.x;
  if (i >= n8) return;
  v4f f0 = *(const v4f*)(in + (size_t)i * 8);
  v4f f1 = *(const v4f*)(in + (size_t)i * 8 + 4);
  v8h o;
#pragma unroll
  for (int t = 0; t < 4; t++) { o[t] = (_Float16)f0[t]; o[4 + t] = (_Float16)f1[t]; }
  *(v8h*)(out + (size_t)i * 8) = o;
}

// ---------------------------------------------------------------------------
// Kernel 1: h = x @ W^T + b for W in {Wq,Wk,Wv} (blockIdx.z selects).
// Block 256 = 8 waves; tile M=128 x N=64, K-step 32, double-buffered LDS
// staging via async-to-LDS (3 x b128 per thread per stage).
// z<2 -> spike = sigmoid(5*(h-1)) thresholded; z==2 -> raw v.  f16 out.
// ---------------------------------------------------------------------------
__global__ __launch_bounds__(256) void proj_kernel(
    const _Float16* __restrict__ xh,
    const _Float16* __restrict__ wqh, const _Float16* __restrict__ wkh,
    const _Float16* __restrict__ wvh,
    const float* __restrict__ bq, const float* __restrict__ bk,
    const float* __restrict__ bv, const float* __restrict__ thr_p,
    _Float16* __restrict__ qf, _Float16* __restrict__ kf, _Float16* __restrict__ vf)
{
  __shared__ _Float16 sX[2][128][32];  // 16 KB
  __shared__ _Float16 sW[2][64][32];   //  8 KB

  const int z = blockIdx.z;
  const _Float16* __restrict__ W    = (z == 0) ? wqh : (z == 1) ? wkh : wvh;
  const float*    __restrict__ bias = (z == 0) ? bq  : (z == 1) ? bk  : bv;
  _Float16*       __restrict__ out  = (z == 0) ? qf  : (z == 1) ? kf  : vf;

  const int tid  = threadIdx.x;
  const int lane = tid & 31;
  const int wave = tid >> 5;
  const int m0   = blockIdx.x * 128;
  const int n0   = blockIdx.y * 64;
  const int lm   = lane & 15;
  const int kbA  = (lane < 16) ? 0 : 8;
  const int kbB  = (lane < 16) ? 0 : 16;

  auto stage = [&](int buf, int kk) {
#pragma unroll
    for (int j = 0; j < 2; j++) {
      const int idx = tid + j * 256;            // 512 b128 for x tile
      const int row = idx >> 2, ch = (idx & 3) * 8;
      async_b128(xh + (size_t)(m0 + row) * D_ + kk + ch, &sX[buf][row][ch]);
    }
    {
      const int row = tid >> 2, ch = (tid & 3) * 8;  // 256 b128 for W tile
      async_b128(W + (size_t)(n0 + row) * D_ + kk + ch, &sW[buf][row][ch]);
    }
  };

  v8f acc[4] = {};
  // issue all fragment loads first, then 4 back-to-back WMMAs
  auto compute = [&](int cur) {
    const _Float16* ax = &sX[cur][wave * 16 + lm][0];
    v16h a  = cat8h(*(const v8h*)(ax + kbB), *(const v8h*)(ax + kbB + 8));
    const _Float16* b0p = &sW[cur][0 * 16 + lm][0];
    const _Float16* b1p = &sW[cur][1 * 16 + lm][0];
    const _Float16* b2p = &sW[cur][2 * 16 + lm][0];
    const _Float16* b3p = &sW[cur][3 * 16 + lm][0];
    v16h b0 = cat8h(*(const v8h*)(b0p + kbB), *(const v8h*)(b0p + kbB + 8));
    v16h b1 = cat8h(*(const v8h*)(b1p + kbB), *(const v8h*)(b1p + kbB + 8));
    v16h b2 = cat8h(*(const v8h*)(b2p + kbB), *(const v8h*)(b2p + kbB + 8));
    v16h b3 = cat8h(*(const v8h*)(b3p + kbB), *(const v8h*)(b3p + kbB + 8));
    acc[0] = WMMA_F16(a, b0, acc[0]);
    acc[1] = WMMA_F16(a, b1, acc[1]);
    acc[2] = WMMA_F16(a, b2, acc[2]);
    acc[3] = WMMA_F16(a, b3, acc[3]);
  };

  stage(0, 0);
  int kk = 0;
  for (; kk + 32 < D_; kk += 32) {          // steady state: branch-free body
    const int cur = (kk >> 5) & 1;
    stage(cur ^ 1, kk + 32);
    WAIT_ASYNC(3);
    __syncthreads();
    compute(cur);
    __syncthreads();
  }
  WAIT_ASYNC(0);                             // peeled last iteration
  __syncthreads();
  compute((kk >> 5) & 1);

  const float thr = *thr_p;
#pragma unroll
  for (int j = 0; j < 4; j++) {
    const int n    = n0 + j * 16 + lm;
    const float bn = bias[n];
#pragma unroll
    for (int r = 0; r < 8; r++) {
      const int m = m0 + wave * 16 + r + ((lane < 16) ? 0 : 8);
      float val = acc[j][r] + bn;
      if (z < 2) {
        float sp = sigf(5.0f * (val - 1.0f));
        val = (sp > thr) ? sp : 0.0f;
      }
      out[(size_t)m * D_ + n] = (_Float16)val;
    }
  }
}

// ---------------------------------------------------------------------------
// Kernel 2: average pool over time, kernel=stride=s.  f32 accumulate.
// ---------------------------------------------------------------------------
__global__ __launch_bounds__(256) void pool_kernel(
    const _Float16* __restrict__ in, _Float16* __restrict__ out, int s, int Ts)
{
  const int idx   = blockIdx.x * blockDim.x + threadIdx.x;
  const int total = B_ * Ts * D_;
  if (idx >= total) return;
  const int c   = idx & (D_ - 1);
  const int row = idx >> 10;
  const int tp  = row % Ts;
  const int b   = row / Ts;
  const _Float16* __restrict__ base = in + (size_t)(b * T_ + tp * s) * D_ + c;
  float acc = 0.0f;
  for (int i = 0; i < s; i++) acc += (float)base[(size_t)i * D_];
  out[idx] = (_Float16)(acc / (float)s);
}

// ---------------------------------------------------------------------------
// Kernel 3: sigmoid attention, one scale.  o = sigmoid(q k^T / 8) @ v.
// grid (Ts/64, B*H); block 128 = 4 waves sharing (b,h).  The 32x64 K and V
// tiles are staged once per block into double-buffered LDS (4 b128/thread),
// so all 4 waves read them from LDS.  Probability tile goes C->A layout via
// a per-wave LDS round trip (DS in-order per wave).
// ---------------------------------------------------------------------------
__global__ __launch_bounds__(128) void attn_kernel(
    const _Float16* __restrict__ q, const _Float16* __restrict__ k,
    const _Float16* __restrict__ v, _Float16* __restrict__ o, int Ts)
{
  __shared__ _Float16 sK[2][32][64];     // 8 KB
  __shared__ _Float16 sV[2][32][64];     // 8 KB
  __shared__ _Float16 ldsW[4][16][32];   // 4 KB

  const int tid  = threadIdx.x;
  const int lane = tid & 31;
  const int wave = tid >> 5;
  const int bh   = blockIdx.y;
  const int b    = bh >> 4;
  const int h    = bh & 15;
  const int rowbase = b * Ts;
  const int qr0  = (blockIdx.x * 4 + wave) * 16;
  const int lm   = lane & 15;
  const int kbA  = (lane < 16) ? 0 : 8;
  const int kbB  = (lane < 16) ? 0 : 16;

  // Q A-fragments (dh 0..31, 32..63), loaded once per wave.
  v16h aq0, aq1;
  {
    const _Float16* __restrict__ qp = q + (size_t)(rowbase + qr0 + lm) * D_ + h * DH_;
    aq0 = cat8h(*(const v8h*)(qp + kbB), *(const v8h*)(qp + kbB + 8));
    aq1 = cat8h(*(const v8h*)(qp + 32 + kbB), *(const v8h*)(qp + 32 + kbB + 8));
  }

  auto stage = [&](int buf, int kt0) {   // 4 async b128 per thread
#pragma unroll
    for (int j = 0; j < 2; j++) {
      const int idx = tid + j * 128;     // 256 b128 per 32x64 tile
      const int row = idx >> 3, ch = (idx & 7) * 8;
      const size_t goff = (size_t)(rowbase + kt0 + row) * D_ + h * DH_ + ch;
      async_b128(k + goff, &sK[buf][row][ch]);
      async_b128(v + goff, &sV[buf][row][ch]);
    }
  };

  const float scale = 0.125f;  // 1/sqrt(DH)
  v8f oacc[4] = {};

  auto compute = [&](int cur) {
    // both 16x16 score tiles: issue all 4 k B-fragment loads, then 4 WMMAs
    const _Float16* kp0 = &sK[cur][0 * 16 + lm][0];
    const _Float16* kp1 = &sK[cur][1 * 16 + lm][0];
    v16h bk00 = cat8h(*(const v8h*)(kp0 + kbB), *(const v8h*)(kp0 + kbB + 8));
    v16h bk01 = cat8h(*(const v8h*)(kp0 + 32 + kbB), *(const v8h*)(kp0 + 32 + kbB + 8));
    v16h bk10 = cat8h(*(const v8h*)(kp1 + kbB), *(const v8h*)(kp1 + kbB + 8));
    v16h bk11 = cat8h(*(const v8h*)(kp1 + 32 + kbB), *(const v8h*)(kp1 + 32 + kbB + 8));
    v8f sc0 = {}, sc1 = {};
    sc0 = WMMA_F16(aq0, bk00, sc0);
    sc0 = WMMA_F16(aq1, bk01, sc0);
    sc1 = WMMA_F16(aq0, bk10, sc1);
    sc1 = WMMA_F16(aq1, bk11, sc1);
    // sigmoid -> per-wave LDS repack buffer (C-layout -> A-layout)
#pragma unroll
    for (int r = 0; r < 8; r++) {
      const int rr = r + ((lane < 16) ? 0 : 8);
      ldsW[wave][rr][0 * 16 + lm] = (_Float16)sigf(sc0[r] * scale);
      ldsW[wave][rr][1 * 16 + lm] = (_Float16)sigf(sc1[r] * scale);
    }
    v16h aw;
#pragma unroll
    for (int i = 0; i < 8; i++) aw[i] = ldsW[wave][lm][kbB + i];
#pragma unroll
    for (int i = 0; i < 8; i++) aw[8 + i] = ldsW[wave][lm][kbB + 8 + i];
    // o += w @ v  (B fragments strided out of LDS)
#pragma unroll
    for (int j = 0; j < 4; j++) {
      v16h bv;
#pragma unroll
      for (int i = 0; i < 16; i++) bv[i] = sV[cur][kbB + i][j * 16 + lm];
      oacc[j] = WMMA_F16(aw, bv, oacc[j]);
    }
  };

  const int npairs = Ts >> 5;
  stage(0, 0);
  int p = 0;
  for (; p + 1 < npairs; p++) {            // steady state: branch-free body
    const int cur = p & 1;
    stage(cur ^ 1, (p + 1) * 32);
    WAIT_ASYNC(4);
    __syncthreads();
    compute(cur);
    __syncthreads();
  }
  WAIT_ASYNC(0);                            // peeled last pair
  __syncthreads();
  compute(p & 1);

#pragma unroll
  for (int j = 0; j < 4; j++) {
    const int n = h * DH_ + j * 16 + lm;
#pragma unroll
    for (int r = 0; r < 8; r++) {
      const int m = qr0 + r + ((lane < 16) ? 0 : 8);
      o[(size_t)(rowbase + m) * D_ + n] = (_Float16)oacc[j][r];
    }
  }
}

// ---------------------------------------------------------------------------
// Kernel 4: out = neuron(cat @ Wo^T + bo).  Upsample+concat folded into A
// addressing (3 region pointers / lane).  Wo tile async-staged to LDS,
// double buffered; A side read directly (f16, L2 resident).
// ---------------------------------------------------------------------------
__global__ __launch_bounds__(256) void final_kernel(
    const _Float16* __restrict__ o1, const _Float16* __restrict__ o3,
    const _Float16* __restrict__ o5, const _Float16* __restrict__ woh,
    const float* __restrict__ bo, float* __restrict__ out)
{
  __shared__ _Float16 sWo[2][64][32];  // 8 KB

  const int tid  = threadIdx.x;
  const int lane = tid & 31;
  const int wave = tid >> 5;
  const int m0   = blockIdx.x * 128 + wave * 16;
  const int n0   = blockIdx.y * 64;
  const int lm   = lane & 15;
  const int kbA  = (lane < 16) ? 0 : 8;
  const int kbB  = (lane < 16) ? 0 : 16;

  const int m = m0 + lm;
  const int b = m / T_;
  const int t = m % T_;
  const _Float16* __restrict__ p0 = o1 + (size_t)m * D_;
  const _Float16* __restrict__ p1 = o3 + (size_t)(b * TS3 + t / 3) * D_;
  const _Float16* __restrict__ p2 = o5 + (size_t)(b * TS5 + t / 5) * D_;

  auto stage = [&](int buf, int kk) {  // 1 async b128 per thread
    const int row = tid >> 2, ch = (tid & 3) * 8;
    async_b128(woh + (size_t)(n0 + row) * (3 * D_) + kk + ch, &sWo[buf][row][ch]);
  };

  v8f acc[4] = {};
  auto compute = [&](int cur, int kk) {
    const int reg = kk >> 10;
    const int cb  = kk & (D_ - 1);
    const _Float16* ap = ((reg == 0) ? p0 : (reg == 1) ? p1 : p2) + cb;
    v16h a = cat8h(*(const v8h*)(ap + kbB), *(const v8h*)(ap + kbB + 8));
    const _Float16* b0p = &sWo[cur][0 * 16 + lm][0];
    const _Float16* b1p = &sWo[cur][1 * 16 + lm][0];
    const _Float16* b2p = &sWo[cur][2 * 16 + lm][0];
    const _Float16* b3p = &sWo[cur][3 * 16 + lm][0];
    v16h b0 = cat8h(*(const v8h*)(b0p + kbB), *(const v8h*)(b0p + kbB + 8));
    v16h b1 = cat8h(*(const v8h*)(b1p + kbB), *(const v8h*)(b1p + kbB + 8));
    v16h b2 = cat8h(*(const v8h*)(b2p + kbB), *(const v8h*)(b2p + kbB + 8));
    v16h b3 = cat8h(*(const v8h*)(b3p + kbB), *(const v8h*)(b3p + kbB + 8));
    acc[0] = WMMA_F16(a, b0, acc[0]);
    acc[1] = WMMA_F16(a, b1, acc[1]);
    acc[2] = WMMA_F16(a, b2, acc[2]);
    acc[3] = WMMA_F16(a, b3, acc[3]);
  };

  stage(0, 0);
  int kk = 0;
  for (; kk + 32 < 3 * D_; kk += 32) {     // steady state: branch-free body
    const int cur = (kk >> 5) & 1;
    stage(cur ^ 1, kk + 32);
    WAIT_ASYNC(1);
    __syncthreads();
    compute(cur, kk);
    __syncthreads();
  }
  WAIT_ASYNC(0);                            // peeled last iteration
  __syncthreads();
  compute((kk >> 5) & 1, kk);

#pragma unroll
  for (int j = 0; j < 4; j++) {
    const int n    = n0 + j * 16 + lm;
    const float bn = bo[n];
#pragma unroll
    for (int r = 0; r < 8; r++) {
      const int mm  = m0 + r + ((lane < 16) ? 0 : 8);
      const float val = acc[j][r] + bn;
      out[(size_t)mm * D_ + n] = sigf(5.0f * (val - 1.0f));
    }
  }
}

// ---------------------------------------------------------------------------
extern "C" void kernel_launch(void* const* d_in, const int* in_sizes, int n_in,
                              void* d_out, int out_size, void* d_ws, size_t ws_size,
                              hipStream_t stream) {
  (void)in_sizes; (void)n_in; (void)out_size; (void)ws_size;

  const float* x   = (const float*)d_in[0];
  const float* Wq  = (const float*)d_in[1];
  const float* bq  = (const float*)d_in[2];
  const float* Wk  = (const float*)d_in[3];
  const float* bk  = (const float*)d_in[4];
  const float* Wv  = (const float*)d_in[5];
  const float* bv  = (const float*)d_in[6];
  const float* Wo  = (const float*)d_in[7];
  const float* bo  = (const float*)d_in[8];
  const float* thr = (const float*)d_in[9];
  float* out = (float*)d_out;

  char* ws = (char*)d_ws;
  size_t off = 0;
  auto take = [&](size_t bytes) -> char* {
    char* p = ws + off;
    off += (bytes + 255) & ~(size_t)255;
    return p;
  };
  _Float16* xh  = (_Float16*)take((size_t)M_ * D_ * 2);
  _Float16* wqh = (_Float16*)take((size_t)D_ * D_ * 2);
  _Float16* wkh = (_Float16*)take((size_t)D_ * D_ * 2);
  _Float16* wvh = (_Float16*)take((size_t)D_ * D_ * 2);
  _Float16* woh = (_Float16*)take((size_t)D_ * 3 * D_ * 2);
  _Float16* qf  = (_Float16*)take((size_t)M_ * D_ * 2);
  _Float16* kf  = (_Float16*)take((size_t)M_ * D_ * 2);
  _Float16* vf  = (_Float16*)take((size_t)M_ * D_ * 2);
  _Float16* qp3 = (_Float16*)take((size_t)B_ * TS3 * D_ * 2);
  _Float16* kp3 = (_Float16*)take((size_t)B_ * TS3 * D_ * 2);
  _Float16* vp3 = (_Float16*)take((size_t)B_ * TS3 * D_ * 2);
  _Float16* qp5 = (_Float16*)take((size_t)B_ * TS5 * D_ * 2);
  _Float16* kp5 = (_Float16*)take((size_t)B_ * TS5 * D_ * 2);
  _Float16* vp5 = (_Float16*)take((size_t)B_ * TS5 * D_ * 2);
  _Float16* o1  = (_Float16*)take((size_t)M_ * D_ * 2);
  _Float16* o3  = (_Float16*)take((size_t)B_ * TS3 * D_ * 2);
  _Float16* o5  = (_Float16*)take((size_t)B_ * TS5 * D_ * 2);

  // 0) one-time f32 -> f16 conversions
  {
    const int nx = M_ * D_ / 8, nw = D_ * D_ / 8, no = D_ * 3 * D_ / 8;
    cvt_kernel<<<dim3((nx + 255) / 256), 256, 0, stream>>>(x, xh, nx);
    cvt_kernel<<<dim3((nw + 255) / 256), 256, 0, stream>>>(Wq, wqh, nw);
    cvt_kernel<<<dim3((nw + 255) / 256), 256, 0, stream>>>(Wk, wkh, nw);
    cvt_kernel<<<dim3((nw + 255) / 256), 256, 0, stream>>>(Wv, wvh, nw);
    cvt_kernel<<<dim3((no + 255) / 256), 256, 0, stream>>>(Wo, woh, no);
  }

  // 1) QKV projections (fused neuron + threshold for q,k)
  proj_kernel<<<dim3(M_ / 128, D_ / 64, 3), 256, 0, stream>>>(
      xh, wqh, wkh, wvh, bq, bk, bv, thr, qf, kf, vf);

  // 2) temporal average pooling for scales 3 and 5
  {
    const int n3 = B_ * TS3 * D_, n5 = B_ * TS5 * D_;
    pool_kernel<<<dim3((n3 + 255) / 256), 256, 0, stream>>>(qf, qp3, 3, TS3);
    pool_kernel<<<dim3((n3 + 255) / 256), 256, 0, stream>>>(kf, kp3, 3, TS3);
    pool_kernel<<<dim3((n3 + 255) / 256), 256, 0, stream>>>(vf, vp3, 3, TS3);
    pool_kernel<<<dim3((n5 + 255) / 256), 256, 0, stream>>>(qf, qp5, 5, TS5);
    pool_kernel<<<dim3((n5 + 255) / 256), 256, 0, stream>>>(kf, kp5, 5, TS5);
    pool_kernel<<<dim3((n5 + 255) / 256), 256, 0, stream>>>(vf, vp5, 5, TS5);
  }

  // 3) sigmoid attention per scale
  attn_kernel<<<dim3(T_ / 64, B_ * H_), 128, 0, stream>>>(qf, kf, vf, o1, T_);
  attn_kernel<<<dim3(TS3 / 64, B_ * H_), 128, 0, stream>>>(qp3, kp3, vp3, o3, TS3);
  attn_kernel<<<dim3(TS5 / 64, B_ * H_), 128, 0, stream>>>(qp5, kp5, vp5, o5, TS5);

  // 4) fused upsample + concat + Wo GEMM + neuron
  final_kernel<<<dim3(M_ / 128, D_ / 64), 256, 0, stream>>>(o1, o3, o5, woh, bo, out);
}
